// AttentionBlock_9792525435528
// MI455X (gfx1250) — matrix operand order repaired and checked by reading, not compiled, not run on hardware
//
#include <hip/hip_runtime.h>
#include <hip/hip_bf16.h>

typedef _Float16 h8   __attribute__((ext_vector_type(8)));
typedef _Float16 h16  __attribute__((ext_vector_type(16)));
typedef float    f8   __attribute__((ext_vector_type(8)));

#define WMMA_F16(A,B,C) __builtin_amdgcn_wmma_f32_16x16x32_f16(false,(A),false,(B),(short)0,(C),false,false)

static __device__ __forceinline__ h16 cat8(h8 lo, h8 hi) {
  return __builtin_shufflevector(lo, hi, 0,1,2,3,4,5,6,7,8,9,10,11,12,13,14,15);
}

// A fragment (16x32 f16). Source row-major [row][k], row stride ld.
// Lane L: M=L%16, half=L/16; elems 0..7 -> K = 8*half+e ; elems 8..15 -> K = 16+8*half+(e-8)
static __device__ __forceinline__ h16 loadA(const _Float16* base, int row, int k0, int ld, int half) {
  const _Float16* p = base + (size_t)row * ld + k0 + 8 * half;
  h8 lo = *(const h8*)p;
  h8 hi = *(const h8*)(p + 16);
  return cat8(lo, hi);
}

// B fragment (32x16 f16). Source stores B^T row-major: elem [n][k], row stride ld.
// Lane L: N=L%16, half=L/16; elems 0..7 -> K = 8*half+e ; elems 8..15 -> K = 16+8*half+(e-8)
static __device__ __forceinline__ h16 loadB(const _Float16* base, int col, int k0, int ld, int half) {
  const _Float16* p = base + (size_t)col * ld + k0 + 8 * half;
  h8 lo = *(const h8*)p;
  h8 hi = *(const h8*)(p + 16);
  return cat8(lo, hi);
}

// ---------------------------------------------------------------- weights f32->f16
__global__ __launch_bounds__(256) void cvt_w_kernel(const float* __restrict__ qkv_w,
                                                    const float* __restrict__ proj_w,
                                                    _Float16* __restrict__ qw16,
                                                    _Float16* __restrict__ pw16) {
  int i = blockIdx.x * 256 + threadIdx.x;
  if (i < 1536 * 512) qw16[i] = (_Float16)qkv_w[i];
  if (i < 512 * 512)  pw16[i] = (_Float16)proj_w[i];
}

// ---------------------------------------------------------------- GroupNorm -> xnT [b][t][c] f16
__global__ __launch_bounds__(256) void gn_kernel(const float* __restrict__ x,
                                                 const float* __restrict__ gw,
                                                 const float* __restrict__ gb,
                                                 _Float16* __restrict__ xnT) {
  int b = blockIdx.x >> 5, g = blockIdx.x & 31;
  const float* xp = x + ((size_t)b * 512 + g * 16) * 1024;   // 16 group channels contiguous
  __shared__ float s1[256], s2[256];
  int tid = threadIdx.x;
  float a = 0.f, q = 0.f;
  for (int i = tid; i < 16384; i += 256) { float v = xp[i]; a += v; q += v * v; }
  s1[tid] = a; s2[tid] = q; __syncthreads();
  for (int off = 128; off > 0; off >>= 1) {
    if (tid < off) { s1[tid] += s1[tid + off]; s2[tid] += s2[tid + off]; }
    __syncthreads();
  }
  float mean = s1[0] * (1.f / 16384.f);
  float var  = s2[0] * (1.f / 16384.f) - mean * mean;
  float rstd = rsqrtf(var + 1e-6f);
  _Float16* dst = xnT + ((size_t)b << 19);
  for (int i = tid; i < 16384; i += 256) {
    int ch = i >> 10, t = i & 1023, c = g * 16 + ch;
    float v = (xp[i] - mean) * rstd * gw[c] + gb[c];
    dst[(size_t)t * 512 + c] = (_Float16)v;
  }
}

// ---------------------------------------------------------------- QKV GEMM (WMMA) -> Qs/Ks [bh][t][64], Vt [bh][64][t]
__global__ __launch_bounds__(128) void qkv_gemm_kernel(const _Float16* __restrict__ qw16,
                                                       const float* __restrict__ qkv_b,
                                                       const _Float16* __restrict__ xnT,
                                                       _Float16* __restrict__ Qs,
                                                       _Float16* __restrict__ Ks,
                                                       _Float16* __restrict__ Vt) {
  int lane = threadIdx.x & 31, wv = threadIdx.x >> 5;
  int m16 = lane & 15, half = lane >> 4;
  int o0 = (blockIdx.x * 4 + wv) * 16;   // 96 o-tiles over 1536
  int t0 = blockIdx.y * 16;              // 64 t-tiles
  int b  = blockIdx.z;
  const _Float16* xb = xnT + ((size_t)b << 19);
  f8 acc = {};
  #pragma unroll 4
  for (int kk = 0; kk < 16; ++kk) {      // K = 512 in steps of 32
    int c0 = kk * 32;
    h16 af = loadA(qw16, o0 + m16, c0, 512, half);
    h16 bf = loadB(xb,   t0 + m16, c0, 512, half);
    acc = WMMA_F16(af, bf, acc);
  }
  // o in [head*192, head*192+192): 0..63=q, 64..127=k, 128..191=v (16-tiles never straddle)
  int head  = o0 / 192;
  int rem   = o0 - head * 192;
  int third = rem / 64;
  int cb    = rem - third * 64;
  int bh    = b * 8 + head;
  int t     = t0 + m16;
  const float scale = 0.3535533905932738f;  // 64^(-1/4), applied to both q and k
  #pragma unroll
  for (int i = 0; i < 8; ++i) {
    int o  = o0 + i + 8 * half;            // D layout: row M = i + 8*(lane/16)
    int cc = cb + i + 8 * half;
    float v = acc[i] + qkv_b[o];
    if (third == 0)      Qs[((size_t)bh << 16) + (size_t)t * 64 + cc]   = (_Float16)(v * scale);
    else if (third == 1) Ks[((size_t)bh << 16) + (size_t)t * 64 + cc]   = (_Float16)(v * scale);
    else                 Vt[((size_t)bh << 16) + (size_t)cc * 1024 + t] = (_Float16)v;
  }
}

// ---------------------------------------------------------------- Flash attention (online softmax, WMMA)
__global__ __launch_bounds__(128) void flash_kernel(const _Float16* __restrict__ Qs,
                                                    const _Float16* __restrict__ Ks,
                                                    const _Float16* __restrict__ Vt,
                                                    _Float16* __restrict__ Arow) {
  int lane = threadIdx.x & 31, wv = threadIdx.x >> 5;
  int m16 = lane & 15, half = lane >> 4;
  int t0 = (blockIdx.x * 4 + wv) * 16;   // 64 query tiles per bh
  int bh = blockIdx.y;
  int b = bh >> 3, head = bh & 7;
  const _Float16* Qb = Qs + ((size_t)bh << 16);
  const _Float16* Kb = Ks + ((size_t)bh << 16);
  const _Float16* Vb = Vt + ((size_t)bh << 16);
  _Float16* Ab = Arow + ((size_t)b << 19) + head * 64;

  __shared__ alignas(16) _Float16 P[4][16][32];   // per-wave D->A layout bounce for probabilities

  h16 qa0 = loadA(Qb, t0 + m16,  0, 64, half);    // Q tile, K=c chunks 0..31 / 32..63
  h16 qa1 = loadA(Qb, t0 + m16, 32, 64, half);

  f8 O0 = {}, O1 = {}, O2 = {}, O3 = {};
  f8 m, l = {};
  #pragma unroll
  for (int i = 0; i < 8; ++i) m[i] = -3.0e38f;

  for (int s0 = 0; s0 < 1024; s0 += 32) {
    f8 S1 = {}, S2 = {};
    S1 = WMMA_F16(qa0, loadB(Kb, s0 + m16,       0, 64, half), S1);
    S1 = WMMA_F16(qa1, loadB(Kb, s0 + m16,      32, 64, half), S1);
    S2 = WMMA_F16(qa0, loadB(Kb, s0 + 16 + m16,  0, 64, half), S2);
    S2 = WMMA_F16(qa1, loadB(Kb, s0 + 16 + m16, 32, 64, half), S2);

    f8 mn, alpha;
    #pragma unroll
    for (int i = 0; i < 8; ++i) {        // row = i + 8*half lives in one 16-lane half
      float v = fmaxf(S1[i], S2[i]);
      v = fmaxf(v, __shfl_xor(v, 1, 32));
      v = fmaxf(v, __shfl_xor(v, 2, 32));
      v = fmaxf(v, __shfl_xor(v, 4, 32));
      v = fmaxf(v, __shfl_xor(v, 8, 32));
      mn[i]    = fmaxf(m[i], v);
      alpha[i] = __expf(m[i] - mn[i]);
      m[i]     = mn[i];
    }
    #pragma unroll
    for (int i = 0; i < 8; ++i) {
      float p1 = __expf(S1[i] - mn[i]);
      float p2 = __expf(S2[i] - mn[i]);
      int row = i + 8 * half;
      P[wv][row][m16]      = (_Float16)p1;
      P[wv][row][16 + m16] = (_Float16)p2;
      float r = p1 + p2;
      r += __shfl_xor(r, 1, 32);
      r += __shfl_xor(r, 2, 32);
      r += __shfl_xor(r, 4, 32);
      r += __shfl_xor(r, 8, 32);
      l[i] = l[i] * alpha[i] + r;
    }
    // re-read P in A-fragment layout (DS ops are in-order within a wave)
    h8 plo = *(const h8*)&P[wv][m16][8 * half];
    h8 phi = *(const h8*)&P[wv][m16][16 + 8 * half];
    h16 pa = cat8(plo, phi);
    #pragma unroll
    for (int i = 0; i < 8; ++i) { O0[i] *= alpha[i]; O1[i] *= alpha[i]; O2[i] *= alpha[i]; O3[i] *= alpha[i]; }
    O0 = WMMA_F16(pa, loadB(Vb,  0 + m16, s0, 1024, half), O0);
    O1 = WMMA_F16(pa, loadB(Vb, 16 + m16, s0, 1024, half), O1);
    O2 = WMMA_F16(pa, loadB(Vb, 32 + m16, s0, 1024, half), O2);
    O3 = WMMA_F16(pa, loadB(Vb, 48 + m16, s0, 1024, half), O3);
  }
  #pragma unroll
  for (int i = 0; i < 8; ++i) {
    float inv = 1.0f / l[i];
    size_t r = (size_t)(t0 + i + 8 * half) * 512;
    Ab[r +      m16] = (_Float16)(O0[i] * inv);
    Ab[r + 16 + m16] = (_Float16)(O1[i] * inv);
    Ab[r + 32 + m16] = (_Float16)(O2[i] * inv);
    Ab[r + 48 + m16] = (_Float16)(O3[i] * inv);
  }
}

// ---------------------------------------------------------------- proj GEMM + residual + 1/sqrt(2)
__global__ __launch_bounds__(128) void proj_kernel(const _Float16* __restrict__ pw16,
                                                   const float* __restrict__ proj_b,
                                                   const _Float16* __restrict__ Arow,
                                                   const float* __restrict__ x,
                                                   float* __restrict__ out) {
  int lane = threadIdx.x & 31, wv = threadIdx.x >> 5;
  int m16 = lane & 15, half = lane >> 4;
  int o0 = (blockIdx.x * 4 + wv) * 16;   // 32 o-tiles over 512
  int t0 = blockIdx.y * 16;
  int b  = blockIdx.z;
  const _Float16* ab = Arow + ((size_t)b << 19);
  f8 acc = {};
  #pragma unroll 4
  for (int kk = 0; kk < 16; ++kk) {
    int c0 = kk * 32;
    h16 af = loadA(pw16, o0 + m16, c0, 512, half);
    h16 bf = loadB(ab,   t0 + m16, c0, 512, half);
    acc = WMMA_F16(af, bf, acc);
  }
  const float inv_sqrt2 = 0.7071067811865476f;
  #pragma unroll
  for (int i = 0; i < 8; ++i) {
    int o = o0 + i + 8 * half;
    size_t off = ((size_t)b * 512 + o) * 1024 + t0 + m16;
    out[off] = (x[off] + acc[i] + proj_b[o]) * inv_sqrt2;
  }
}

extern "C" void kernel_launch(void* const* d_in, const int* in_sizes, int n_in,
                              void* d_out, int out_size, void* d_ws, size_t ws_size,
                              hipStream_t stream) {
  (void)in_sizes; (void)n_in; (void)out_size; (void)ws_size;
  const float* x      = (const float*)d_in[0];
  const float* gn_w   = (const float*)d_in[1];
  const float* gn_b   = (const float*)d_in[2];
  const float* qkv_w  = (const float*)d_in[3];
  const float* qkv_b  = (const float*)d_in[4];
  const float* proj_w = (const float*)d_in[5];
  const float* proj_b = (const float*)d_in[6];
  float* out = (float*)d_out;

  char* ws = (char*)d_ws;                       // total use: ~42.5 MB
  _Float16* xnT  = (_Float16*)(ws);             // [8][1024][512]  8 MB
  _Float16* Qs   = (_Float16*)(ws + ( 8u << 20)); // [64][1024][64] 8 MB (scaled)
  _Float16* Ks   = (_Float16*)(ws + (16u << 20)); // [64][1024][64] 8 MB (scaled)
  _Float16* Vt   = (_Float16*)(ws + (24u << 20)); // [64][64][1024] 8 MB
  _Float16* Arow = (_Float16*)(ws + (32u << 20)); // [8][1024][512] 8 MB
  _Float16* qw16 = (_Float16*)(ws + (40u << 20)); // 1.5 MB
  _Float16* pw16 = (_Float16*)(ws + (42u << 20)); // 0.5 MB

  cvt_w_kernel<<<3072, 256, 0, stream>>>(qkv_w, proj_w, qw16, pw16);
  gn_kernel<<<256, 256, 0, stream>>>(x, gn_w, gn_b, xnT);
  qkv_gemm_kernel<<<dim3(24, 64, 8), 128, 0, stream>>>(qw16, qkv_b, xnT, Qs, Ks, Vt);
  flash_kernel<<<dim3(16, 64), 128, 0, stream>>>(Qs, Ks, Vt, Arow);
  proj_kernel<<<dim3(8, 64, 8), 128, 0, stream>>>(pw16, proj_b, Arow, x, out);
}
